// AnticipationModel_33054068310483
// MI455X (gfx1250) — matrix operand order repaired and checked by reading, not compiled, not run on hardware
//
#include <hip/hip_runtime.h>
#include <math.h>
#include <stdint.h>

// ---------------------------------------------------------------------------
// VRNN-style anticipation model on gfx1250 (MI455X).
//  - all GEMMs via v_wmma_f32_16x16x32_f16 (f16 in, f32 accum)
//  - one wave32 per 16 batch rows; wave privately owns its recurrence
//  - weights pre-converted to f16 transposed (output-major) layout in d_ws
//  - phi_act precomputed as a 20x64 f16 table (branch-free A-frag gather)
//  - D->A layout transposes through per-wave LDS scratch (in-order DS pipe)
// ---------------------------------------------------------------------------

typedef _Float16 h16;
typedef __attribute__((ext_vector_type(16))) _Float16 v16h;
typedef __attribute__((ext_vector_type(8)))  _Float16 v8h;
typedef __attribute__((ext_vector_type(8)))  float    v8f;

#define LDS_FENCE() asm volatile("s_wait_dscnt 0" ::: "memory")

constexpr int ACT_DIM = 20, H = 64, ZD = 64, NZ = 5, Bn = 1024, Tn = 128, NSn = 1000;

// workspace layout (in halves), all weights transposed to [out][in], f16
constexpr int OFF_WXT    = 0;                        // [64][128]
constexpr int OFF_WZT    = OFF_WXT    + 64 * 128;    // [64][64]
constexpr int OFF_WP1T   = OFF_WZT    + 64 * 64;     // [5][64][128]
constexpr int OFF_WP2T   = OFF_WP1T   + 5 * 64 * 128;// [5][64][64]
constexpr int OFF_WQ1T   = OFF_WP2T   + 5 * 64 * 64; // [5][64][64]
constexpr int OFF_WQ2T   = OFF_WQ1T   + 5 * 64 * 64; // [5][64][64]
constexpr int OFF_WIHT   = OFF_WQ2T   + 5 * 64 * 64; // [192][128]
constexpr int OFF_WHHT   = OFF_WIHT   + 192 * 128;   // [192][64]
constexpr int OFF_WDECT  = OFF_WHHT   + 192 * 64;    // [64][128]
constexpr int OFF_WAT    = OFF_WDECT  + 64 * 128;    // [32][64] (N padded 20->32)
constexpr int OFF_WACTT  = OFF_WAT    + 32 * 64;     // [64][32] (K padded 20->32)
constexpr int OFF_PHIACT = OFF_WACTT  + 64 * 32;     // [20][64] relu(W_act+b_act)
constexpr int WS_TOTAL   = OFF_PHIACT + ACT_DIM * 64;

constexpr int OUT_DUR = Bn * ACT_DIM;   // 20480
constexpr int OUT_SP  = OUT_DUR + Bn;   // 21504
constexpr int OUT_KLD = OUT_SP + Bn;    // 22528

// ---------------------------------------------------------------------------
// fragment helpers (16x16x32 f16 WMMA)
// A (16x32): lane<16 -> M=lane, K = 0..7 & 16..23 ; lane>=16 -> K = 8..15 & 24..31
// B (32x16): lane<16 -> N=lane, K = 0..7 & 16..23 ; lane>=16 -> N=lane-16, K = 8..15 & 24..31
//            -> contiguous if weights stored transposed [N][K]
// C/D (16x16 f32): vgpr r -> M = r + 8*(lane>>4), N = lane&15
// ---------------------------------------------------------------------------
__device__ inline v8f wmma(v16h a, v16h b, v8f c) {
  return __builtin_amdgcn_wmma_f32_16x16x32_f16(false, a, false, b, (short)0, c,
                                                false, false);
}

__device__ inline v16h cat8(v8h lo, v8h hi8) {
  v16h a;
#pragma unroll
  for (int i = 0; i < 8; ++i) { a[i] = lo[i]; a[i + 8] = hi8[i]; }
  return a;
}

__device__ inline v16h load_a(const h16* base, int ld, int k0, int lane) {
  const h16* r = base + (lane & 15) * ld + k0 + ((lane >> 4) << 3);
  return cat8(*(const v8h*)r, *(const v8h*)(r + 16));
}

__device__ inline v16h load_b(const h16* Wt, int ldk, int n0, int k0, int lane) {
  const h16* r = Wt + (n0 + (lane & 15)) * ldk + k0 + ((lane >> 4) << 3);
  v16h b;
#pragma unroll
  for (int i = 0; i < 16; ++i) b[i] = r[i < 8 ? i : i + 8];
  return b;
}

// fast elementwise helpers (single-trans-op exp/log)
__device__ inline float softplusf(float x) {
  return (x > 20.f) ? x : __logf(1.f + __expf(x));
}
__device__ inline float sigmoidf_(float x) { return 1.f / (1.f + __expf(-x)); }
__device__ inline float tanhf_(float x) {
  float e = __expf(2.f * x);
  return (e - 1.f) / (e + 1.f);
}

// Counter-hash Gaussian. NOTE: the reference averages NS=1000 JAX threefry
// normals (eps ~ N(0, 1/NS)); bit-exact threefry reproduction is out of scope
// here, so we generate a distributionally-equivalent eps from a hash.
__device__ inline uint32_t mix32(uint32_t x) {
  x ^= x >> 16; x *= 0x7feb352du; x ^= x >> 15; x *= 0x846ca68bu; x ^= x >> 16;
  return x;
}
__device__ inline float hash_gauss(uint32_t t, uint32_t b, uint32_t z) {
  uint32_t h1 = mix32(t * 0x9E3779B9u ^ b * 0x85EBCA6Bu ^ z * 0xC2B2AE35u ^ 0xA511E9B3u);
  uint32_t h2 = mix32(h1 ^ 0x63D83595u);
  float u1 = fmaxf((h1 >> 8) * 5.9604645e-8f, 1.1920929e-7f);
  float u2 = (h2 >> 8) * 5.9604645e-8f;
  return sqrtf(-2.f * __logf(u1)) * __cosf(6.28318530718f * u2);
}

// ---------------------------------------------------------------------------
// 2-layer 5-way mixture with 64-dim input (used for prior; post is inlined):
//   out = sum_n gamma[n] * ( relu(A @ W1_n + b1_n) @ W2_n + b2_n )
// ---------------------------------------------------------------------------
__device__ inline void mixture64(const h16* W1base, const float* b1,
                                 const h16* W2base, const float* b2,
                                 const float* gamma, v16h a0, v16h a1,
                                 h16* scr, int lane, v8f outacc[4]) {
  const int colL = lane & 15, hi = lane >> 4;
#pragma unroll
  for (int nt = 0; nt < 4; ++nt) outacc[nt] = {};
  for (int n = 0; n < NZ; ++n) {
    const h16* W1 = W1base + n * 64 * 64;
#pragma unroll
    for (int nt = 0; nt < 4; ++nt) {
      v8f acc = {};
      acc = wmma(a0, load_b(W1, 64, nt * 16, 0, lane), acc);
      acc = wmma(a1, load_b(W1, 64, nt * 16, 32, lane), acc);
      float bv = b1[n * 64 + nt * 16 + colL];
#pragma unroll
      for (int r = 0; r < 8; ++r)
        scr[(r + 8 * hi) * 64 + nt * 16 + colL] = (h16)fmaxf(acc[r] + bv, 0.f);
    }
    LDS_FENCE();
    v16h m0 = load_a(scr, 64, 0, lane), m1 = load_a(scr, 64, 32, lane);
    const h16* W2 = W2base + n * 64 * 64;
    float g = gamma[n];
#pragma unroll
    for (int nt = 0; nt < 4; ++nt) {
      v8f acc = {};
      acc = wmma(m0, load_b(W2, 64, nt * 16, 0, lane), acc);
      acc = wmma(m1, load_b(W2, 64, nt * 16, 32, lane), acc);
      float bv = b2[n * 64 + nt * 16 + colL];
#pragma unroll
      for (int r = 0; r < 8; ++r) outacc[nt][r] += g * (acc[r] + bv);
    }
  }
}

// ---------------------------------------------------------------------------
// prep: convert + transpose all weights to f16 [out][in]; build phi_act table
// ---------------------------------------------------------------------------
__global__ void vrnn_prep(const float* __restrict__ W_x, const float* __restrict__ W_z,
                          const float* __restrict__ Wp1, const float* __restrict__ Wp2,
                          const float* __restrict__ Wq1, const float* __restrict__ Wq2,
                          const float* __restrict__ W_ih, const float* __restrict__ W_hh,
                          const float* __restrict__ W_dec, const float* __restrict__ W_a,
                          const float* __restrict__ W_act, const float* __restrict__ b_act,
                          h16* __restrict__ ws, float* __restrict__ out) {
  int i = blockIdx.x * blockDim.x + threadIdx.x;
  if (i == 0) out[OUT_KLD] = 0.f;
  if (i >= WS_TOTAL) return;
  float v;
  if (i < OFF_WZT) {
    int j = i - OFF_WXT, o = j / 128, k = j % 128;
    v = W_x[k * 64 + o];
  } else if (i < OFF_WP1T) {
    int j = i - OFF_WZT, o = j / 64, k = j % 64;
    v = W_z[k * 64 + o];
  } else if (i < OFF_WP2T) {
    int j = i - OFF_WP1T, n = j / (64 * 128), rem = j % (64 * 128), o = rem / 128, k = rem % 128;
    v = Wp1[(n * 128 + k) * 64 + o];
  } else if (i < OFF_WQ1T) {
    int j = i - OFF_WP2T, n = j / 4096, rem = j % 4096, o = rem / 64, k = rem % 64;
    v = Wp2[(n * 64 + k) * 64 + o];
  } else if (i < OFF_WQ2T) {
    int j = i - OFF_WQ1T, n = j / 4096, rem = j % 4096, o = rem / 64, k = rem % 64;
    v = Wq1[(n * 64 + k) * 64 + o];
  } else if (i < OFF_WIHT) {
    int j = i - OFF_WQ2T, n = j / 4096, rem = j % 4096, o = rem / 64, k = rem % 64;
    v = Wq2[(n * 64 + k) * 64 + o];
  } else if (i < OFF_WHHT) {
    int j = i - OFF_WIHT, o = j / 128, k = j % 128;
    v = W_ih[k * 192 + o];
  } else if (i < OFF_WDECT) {
    int j = i - OFF_WHHT, o = j / 64, k = j % 64;
    v = W_hh[k * 192 + o];
  } else if (i < OFF_WAT) {
    int j = i - OFF_WDECT, o = j / 128, k = j % 128;
    v = W_dec[k * 64 + o];
  } else if (i < OFF_WACTT) {
    int j = i - OFF_WAT, o = j / 64, k = j % 64;
    v = (o < ACT_DIM) ? W_a[k * ACT_DIM + o] : 0.f;
  } else if (i < OFF_PHIACT) {
    int j = i - OFF_WACTT, o = j / 32, k = j % 32;
    v = (k < ACT_DIM) ? W_act[k * 64 + o] : 0.f;
  } else {
    int j = i - OFF_PHIACT, a = j / 64, k = j % 64;   // row-major [act][k]
    v = fmaxf(W_act[a * 64 + k] + b_act[k], 0.f);
  }
  ws[i] = (h16)v;
}

// ---------------------------------------------------------------------------
// main scan: one wave32 per 16 batch rows, full T-loop per wave
// ---------------------------------------------------------------------------
__global__ __launch_bounds__(32) void vrnn_scan(
    const int* __restrict__ acts, const float* __restrict__ durs,
    const float* __restrict__ b_act, const float* __restrict__ W_dur,
    const float* __restrict__ b_dur, const float* __restrict__ b_x,
    const float* __restrict__ b_z, const float* __restrict__ bp1,
    const float* __restrict__ bp2, const float* __restrict__ bq1,
    const float* __restrict__ bq2, const float* __restrict__ b_dec,
    const float* __restrict__ b_a, const float* __restrict__ W_durd,
    const float* __restrict__ b_durd, const float* __restrict__ b_ih,
    const float* __restrict__ b_hh, const float* __restrict__ gpost,
    const float* __restrict__ gprior, const h16* __restrict__ ws,
    float* __restrict__ out) {
  __shared__ h16 scr[16 * 64];   // per-wave transpose staging
  __shared__ h16 hbuf[16 * 64];  // f16 copy of h (and later dec)

  const int lane = threadIdx.x;       // 0..31 (one wave per block)
  const int colL = lane & 15;
  const int hi   = lane >> 4;
  const int kb   = hi << 3;           // A-frag K sub-offset: 0 or 8
  const int row0 = blockIdx.x * 16;   // this wave's batch rows

  const h16* Wxt    = ws + OFF_WXT;
  const h16* Wzt    = ws + OFF_WZT;
  const h16* Wp1t   = ws + OFF_WP1T;
  const h16* Wq1t   = ws + OFF_WQ1T;
  const h16* Wp2t   = ws + OFF_WP2T;
  const h16* Wq2t   = ws + OFF_WQ2T;
  const h16* Wiht   = ws + OFF_WIHT;
  const h16* Whht   = ws + OFF_WHHT;
  const h16* Wdect  = ws + OFF_WDECT;
  const h16* Wat    = ws + OFF_WAT;
  const h16* Wactt  = ws + OFF_WACTT;
  const h16* phiTab = ws + OFF_PHIACT;

  const float invSqrtNS = rsqrtf((float)NSn);

  v8f hreg[4];                       // recurrent state, D-layout f32, cols 0..63
#pragma unroll
  for (int nt = 0; nt < 4; ++nt) hreg[nt] = {};
  float kacc = 0.f;                  // per-lane partial of sum_b sum_z sum_t kl

  for (int t = 0; t < Tn; ++t) {
    // ---- build A-frags of [phi_act | phi_dur] (K=128), branch-free ----
    const int   act_m = acts[(row0 + colL) * Tn + t];
    const float dur_m = durs[(row0 + colL) * Tn + t];
    if (t + 1 < Tn) {
      __builtin_prefetch(&acts[(row0 + colL) * Tn + t + 1], 0, 3);
      __builtin_prefetch(&durs[(row0 + colL) * Tn + t + 1], 0, 3);
    }
    v16h axin[4];
    {
      const h16* arow = phiTab + act_m * 64;   // phi_act row (f16 table)
#pragma unroll
      for (int kt = 0; kt < 2; ++kt) {
        const h16* r = arow + kt * 32 + kb;
        axin[kt] = cat8(*(const v8h*)r, *(const v8h*)(r + 16));
      }
#pragma unroll
      for (int kt = 0; kt < 2; ++kt) {        // phi_dur half, uniform offsets
        v16h a;
#pragma unroll
        for (int c = 0; c < 2; ++c) {
          const int kk = kt * 32 + kb + c * 16;
          float4 w0 = *(const float4*)(W_dur + kk);
          float4 w1 = *(const float4*)(W_dur + kk + 4);
          float4 c0 = *(const float4*)(b_dur + kk);
          float4 c1 = *(const float4*)(b_dur + kk + 4);
          a[c * 8 + 0] = (h16)fmaxf(dur_m * w0.x + c0.x, 0.f);
          a[c * 8 + 1] = (h16)fmaxf(dur_m * w0.y + c0.y, 0.f);
          a[c * 8 + 2] = (h16)fmaxf(dur_m * w0.z + c0.z, 0.f);
          a[c * 8 + 3] = (h16)fmaxf(dur_m * w0.w + c0.w, 0.f);
          a[c * 8 + 4] = (h16)fmaxf(dur_m * w1.x + c1.x, 0.f);
          a[c * 8 + 5] = (h16)fmaxf(dur_m * w1.y + c1.y, 0.f);
          a[c * 8 + 6] = (h16)fmaxf(dur_m * w1.z + c1.z, 0.f);
          a[c * 8 + 7] = (h16)fmaxf(dur_m * w1.w + c1.w, 0.f);
        }
        axin[2 + kt] = a;
      }
    }

    // ---- phi_x = relu([phi_act | phi_dur] @ W_x + b_x) ----
    v8f px[4];
#pragma unroll
    for (int nt = 0; nt < 4; ++nt) px[nt] = {};
#pragma unroll
    for (int kt = 0; kt < 4; ++kt)
#pragma unroll
      for (int nt = 0; nt < 4; ++nt)
        px[nt] = wmma(axin[kt], load_b(Wxt, 128, nt * 16, kt * 32, lane), px[nt]);

    // stage phi_x (f16) and h (f16) for A-fragment reload
#pragma unroll
    for (int nt = 0; nt < 4; ++nt) {
      float bv = b_x[nt * 16 + colL];
#pragma unroll
      for (int r = 0; r < 8; ++r) {
        scr[(r + 8 * hi) * 64 + nt * 16 + colL]  = (h16)fmaxf(px[nt][r] + bv, 0.f);
        hbuf[(r + 8 * hi) * 64 + nt * 16 + colL] = (h16)hreg[nt][r];
      }
    }
    LDS_FENCE();
    v16h ax0 = load_a(scr, 64, 0, lane),  ax1 = load_a(scr, 64, 32, lane);
    v16h ah0 = load_a(hbuf, 64, 0, lane), ah1 = load_a(hbuf, 64, 32, lane);

    // ---- posterior mixture: input = [phi_x | h] (K=128) ----
    v8f macc[4];
#pragma unroll
    for (int nt = 0; nt < 4; ++nt) macc[nt] = {};
    for (int n = 0; n < NZ; ++n) {
      const h16* W1 = Wp1t + n * 64 * 128;
#pragma unroll
      for (int nt = 0; nt < 4; ++nt) {
        v8f acc = {};
        acc = wmma(ax0, load_b(W1, 128, nt * 16, 0, lane), acc);
        acc = wmma(ax1, load_b(W1, 128, nt * 16, 32, lane), acc);
        acc = wmma(ah0, load_b(W1, 128, nt * 16, 64, lane), acc);
        acc = wmma(ah1, load_b(W1, 128, nt * 16, 96, lane), acc);
        float bv = bp1[n * 64 + nt * 16 + colL];
#pragma unroll
        for (int r = 0; r < 8; ++r)
          scr[(r + 8 * hi) * 64 + nt * 16 + colL] = (h16)fmaxf(acc[r] + bv, 0.f);
      }
      LDS_FENCE();
      v16h m0 = load_a(scr, 64, 0, lane), m1 = load_a(scr, 64, 32, lane);
      const h16* W2 = Wp2t + n * 64 * 64;
      float g = gpost[n];
#pragma unroll
      for (int nt = 0; nt < 4; ++nt) {
        v8f acc = {};
        acc = wmma(m0, load_b(W2, 64, nt * 16, 0, lane), acc);
        acc = wmma(m1, load_b(W2, 64, nt * 16, 32, lane), acc);
        float bv = bp2[n * 64 + nt * 16 + colL];
#pragma unroll
        for (int r = 0; r < 8; ++r) macc[nt][r] += g * (acc[r] + bv);
      }
    }

    // ---- prior mixture: input = h (K=64) ----
    v8f qacc[4];
    mixture64(Wq1t, bq1, Wq2t, bq2, gprior, ah0, ah1, scr, lane, qacc);

    // ---- GRU gate pre-activations from h: gh (r,z) and hn ----
    v8f accg[12], acchn[4];
#pragma unroll
    for (int nt = 0; nt < 8; ++nt) {
      v8f acc = {};
      acc = wmma(ah0, load_b(Whht, 64, nt * 16, 0, lane), acc);
      acc = wmma(ah1, load_b(Whht, 64, nt * 16, 32, lane), acc);
      accg[nt] = acc;
    }
#pragma unroll
    for (int nt = 0; nt < 4; ++nt) {
      v8f acc = {};
      acc = wmma(ah0, load_b(Whht, 64, (8 + nt) * 16, 0, lane), acc);
      acc = wmma(ah1, load_b(Whht, 64, (8 + nt) * 16, 32, lane), acc);
      acchn[nt] = acc;
      v8f z0 = {};
      accg[8 + nt] = z0;
    }

    // ---- KL + sample z (elementwise, D-layout) ----
#pragma unroll
    for (int nt = 0; nt < 4; ++nt) {
#pragma unroll
      for (int r = 0; r < 8; ++r) {
        float m = macc[nt][r], q = qacc[nt][r];
        float pm = fmaxf(m, 0.f), ps = softplusf(m);
        float qm = fmaxf(q, 0.f), qs = softplusf(q);
        float d = pm - qm;
        kacc += __logf(qs / ps) + (ps * ps + d * d) / (2.f * qs * qs) - 0.5f;
        float eps = hash_gauss((uint32_t)t, (uint32_t)(row0 + r + 8 * hi),
                               (uint32_t)(nt * 16 + colL)) * invSqrtNS;
        scr[(r + 8 * hi) * 64 + nt * 16 + colL] = (h16)(pm + ps * eps);
      }
    }
    LDS_FENCE();
    v16h az0 = load_a(scr, 64, 0, lane), az1 = load_a(scr, 64, 32, lane);

    // ---- phi_z = relu(z @ W_z + b_z) ----
#pragma unroll
    for (int nt = 0; nt < 4; ++nt) {
      v8f acc = {};
      acc = wmma(az0, load_b(Wzt, 64, nt * 16, 0, lane), acc);
      acc = wmma(az1, load_b(Wzt, 64, nt * 16, 32, lane), acc);
      float bv = b_z[nt * 16 + colL];
#pragma unroll
      for (int r = 0; r < 8; ++r)
        scr[(r + 8 * hi) * 64 + nt * 16 + colL] = (h16)fmaxf(acc[r] + bv, 0.f);
    }
    LDS_FENCE();
    v16h apz0 = load_a(scr, 64, 0, lane), apz1 = load_a(scr, 64, 32, lane);

    // ---- gi = [phi_x | phi_z] @ W_ih, accumulated onto gh for r,z gates ----
#pragma unroll
    for (int nt = 0; nt < 12; ++nt) {
      v8f acc = accg[nt];
      acc = wmma(ax0,  load_b(Wiht, 128, nt * 16, 0, lane), acc);
      acc = wmma(ax1,  load_b(Wiht, 128, nt * 16, 32, lane), acc);
      acc = wmma(apz0, load_b(Wiht, 128, nt * 16, 64, lane), acc);
      acc = wmma(apz1, load_b(Wiht, 128, nt * 16, 96, lane), acc);
      accg[nt] = acc;
    }

    // ---- GRU elementwise update ----
#pragma unroll
    for (int nt = 0; nt < 4; ++nt) {
      int col = nt * 16 + colL;
      float bir = b_ih[col] + b_hh[col];
      float biz = b_ih[64 + col] + b_hh[64 + col];
      float bin = b_ih[128 + col], bhn = b_hh[128 + col];
#pragma unroll
      for (int r = 0; r < 8; ++r) {
        float rg = sigmoidf_(accg[nt][r] + bir);
        float zg = sigmoidf_(accg[4 + nt][r] + biz);
        float nn = tanhf_(accg[8 + nt][r] + bin + rg * (acchn[nt][r] + bhn));
        hreg[nt][r] = (1.f - zg) * nn + zg * hreg[nt][r];
      }
    }
  }  // t loop

  // ======================= epilogue (decoder head) =========================
  // stage final h
#pragma unroll
  for (int nt = 0; nt < 4; ++nt)
#pragma unroll
    for (int r = 0; r < 8; ++r)
      hbuf[(r + 8 * hi) * 64 + nt * 16 + colL] = (h16)hreg[nt][r];
  LDS_FENCE();
  v16h ah0 = load_a(hbuf, 64, 0, lane), ah1 = load_a(hbuf, 64, 32, lane);

  v8f qacc[4];
  mixture64(Wq1t, bq1, Wq2t, bq2, gprior, ah0, ah1, scr, lane, qacc);
#pragma unroll
  for (int nt = 0; nt < 4; ++nt)
#pragma unroll
    for (int r = 0; r < 8; ++r) {
      float q = qacc[nt][r];
      float qm = fmaxf(q, 0.f), qs = softplusf(q);
      float eps = hash_gauss((uint32_t)Tn, (uint32_t)(row0 + r + 8 * hi),
                             (uint32_t)(nt * 16 + colL)) * invSqrtNS;
      scr[(r + 8 * hi) * 64 + nt * 16 + colL] = (h16)(qm + qs * eps);
    }
  LDS_FENCE();
  v16h az0 = load_a(scr, 64, 0, lane), az1 = load_a(scr, 64, 32, lane);
#pragma unroll
  for (int nt = 0; nt < 4; ++nt) {
    v8f acc = {};
    acc = wmma(az0, load_b(Wzt, 64, nt * 16, 0, lane), acc);
    acc = wmma(az1, load_b(Wzt, 64, nt * 16, 32, lane), acc);
    float bv = b_z[nt * 16 + colL];
#pragma unroll
    for (int r = 0; r < 8; ++r)
      scr[(r + 8 * hi) * 64 + nt * 16 + colL] = (h16)fmaxf(acc[r] + bv, 0.f);
  }
  LDS_FENCE();
  v16h apz0 = load_a(scr, 64, 0, lane), apz1 = load_a(scr, 64, 32, lane);

  // dec = relu([phi_z | h] @ W_dec + b_dec) -> stage into hbuf
#pragma unroll
  for (int nt = 0; nt < 4; ++nt) {
    v8f acc = {};
    acc = wmma(apz0, load_b(Wdect, 128, nt * 16, 0, lane), acc);
    acc = wmma(apz1, load_b(Wdect, 128, nt * 16, 32, lane), acc);
    acc = wmma(ah0,  load_b(Wdect, 128, nt * 16, 64, lane), acc);
    acc = wmma(ah1,  load_b(Wdect, 128, nt * 16, 96, lane), acc);
    float bv = b_dec[nt * 16 + colL];
#pragma unroll
    for (int r = 0; r < 8; ++r)
      hbuf[(r + 8 * hi) * 64 + nt * 16 + colL] = (h16)fmaxf(acc[r] + bv, 0.f);
  }
  LDS_FENCE();
  v16h ad0 = load_a(hbuf, 64, 0, lane), ad1 = load_a(hbuf, 64, 32, lane);

  // logits = dec @ W_a + b_a (N padded to 32); write out + stage f16 padded
#pragma unroll
  for (int nt = 0; nt < 2; ++nt) {
    v8f acc = {};
    acc = wmma(ad0, load_b(Wat, 64, nt * 16, 0, lane), acc);
    acc = wmma(ad1, load_b(Wat, 64, nt * 16, 32, lane), acc);
    int col = nt * 16 + colL;
    float bv = (col < ACT_DIM) ? b_a[col] : 0.f;
#pragma unroll
    for (int r = 0; r < 8; ++r) {
      float v = acc[r] + bv;
      int grow = row0 + r + 8 * hi;
      if (col < ACT_DIM) out[grow * ACT_DIM + col] = v;
      scr[(r + 8 * hi) * 32 + col] = (h16)((col < ACT_DIM) ? v : 0.f);
    }
  }
  LDS_FENCE();
  v16h alog = load_a(scr, 32, 0, lane);

  // phi_logits = relu(logits @ W_act + b_act) -> stage [16][64] in scr
  {
    v8f pl[4];
#pragma unroll
    for (int nt = 0; nt < 4; ++nt) {
      v8f acc = {};
      acc = wmma(alog, load_b(Wactt, 32, nt * 16, 0, lane), acc);
      pl[nt] = acc;
    }
    LDS_FENCE();  // alog consumed before overwriting scr
#pragma unroll
    for (int nt = 0; nt < 4; ++nt) {
      float bv = b_act[nt * 16 + colL];
#pragma unroll
      for (int r = 0; r < 8; ++r)
        scr[(r + 8 * hi) * 64 + nt * 16 + colL] = (h16)fmaxf(pl[nt][r] + bv, 0.f);
    }
    LDS_FENCE();
  }

  // dur_dec = [phi_logits | dec] @ W_durd + b_durd  (N=1, serial per row)
  if (lane < 16) {
    int grow = row0 + lane;
    float s = b_durd[0];
#pragma unroll 8
    for (int k = 0; k < 64; ++k) {
      s += (float)scr[lane * 64 + k] * W_durd[k];
      s += (float)hbuf[lane * 64 + k] * W_durd[64 + k];
    }
    out[OUT_DUR + grow] = s;
    out[OUT_SP + grow]  = softplusf(s);
  }

  // mean(kld): total over all (b,z,t) / B
  float s = kacc;
#pragma unroll
  for (int off = 16; off > 0; off >>= 1) s += __shfl_xor(s, off, 32);
  if (lane == 0) atomicAdd(&out[OUT_KLD], s * (1.0f / (float)Bn));
}

// ---------------------------------------------------------------------------
extern "C" void kernel_launch(void* const* d_in, const int* in_sizes, int n_in,
                              void* d_out, int out_size, void* d_ws, size_t ws_size,
                              hipStream_t stream) {
  const int*   acts   = (const int*)d_in[0];
  const float* durs   = (const float*)d_in[1];
  const float* W_act  = (const float*)d_in[2];
  const float* b_act  = (const float*)d_in[3];
  const float* W_dur  = (const float*)d_in[4];
  const float* b_dur  = (const float*)d_in[5];
  const float* W_x    = (const float*)d_in[6];
  const float* b_x    = (const float*)d_in[7];
  const float* W_z    = (const float*)d_in[8];
  const float* b_z    = (const float*)d_in[9];
  const float* Wp1    = (const float*)d_in[10];
  const float* bp1    = (const float*)d_in[11];
  const float* Wp2    = (const float*)d_in[12];
  const float* bp2    = (const float*)d_in[13];
  const float* Wq1    = (const float*)d_in[14];
  const float* bq1    = (const float*)d_in[15];
  const float* Wq2    = (const float*)d_in[16];
  const float* bq2    = (const float*)d_in[17];
  const float* W_dec  = (const float*)d_in[18];
  const float* b_dec  = (const float*)d_in[19];
  const float* W_a    = (const float*)d_in[20];
  const float* b_a    = (const float*)d_in[21];
  const float* W_durd = (const float*)d_in[22];
  const float* b_durd = (const float*)d_in[23];
  const float* W_ih   = (const float*)d_in[24];
  const float* W_hh   = (const float*)d_in[25];
  const float* b_ih   = (const float*)d_in[26];
  const float* b_hh   = (const float*)d_in[27];
  const float* gpost  = (const float*)d_in[28];
  const float* gprior = (const float*)d_in[29];

  h16*   ws  = (h16*)d_ws;
  float* out = (float*)d_out;

  vrnn_prep<<<(WS_TOTAL + 255) / 256, 256, 0, stream>>>(
      W_x, W_z, Wp1, Wp2, Wq1, Wq2, W_ih, W_hh, W_dec, W_a, W_act, b_act, ws, out);

  vrnn_scan<<<Bn / 16, 32, 0, stream>>>(
      acts, durs, b_act, W_dur, b_dur, b_x, b_z, bp1, bp2, bq1, bq2,
      b_dec, b_a, W_durd, b_durd, b_ih, b_hh, gpost, gprior, ws, out);
}
